// Model_MSE_53592601919728
// MI455X (gfx1250) — matrix operand order repaired and checked by reading, not compiled, not run on hardware
//
#include <hip/hip_runtime.h>
#include <hip/hip_bf16.h>

// Problem constants (from reference)
#define BB   128
#define LL   4096
#define EE   50
#define FF   100
#define HH   250
#define KSEL 64
#define CP   128   // padded channel count for all conv activations

typedef __attribute__((ext_vector_type(16))) __bf16 v16bf;
typedef __attribute__((ext_vector_type(8)))  float  v8f;

struct U256 { uint4 lo, hi; };

static __device__ __forceinline__ v16bf mkfrag(uint4 lo, uint4 hi) {
  U256 u; u.lo = lo; u.hi = hi;
  return __builtin_bit_cast(v16bf, u);
}

static __device__ __forceinline__ unsigned short f2bf(float f) {
  unsigned u = __float_as_uint(f);
  u += 0x7FFFu + ((u >> 16) & 1u);       // round-to-nearest-even
  return (unsigned short)(u >> 16);
}
static __device__ __forceinline__ float bf2f(unsigned short h) {
  return __uint_as_float(((unsigned)h) << 16);
}

// Branch-free zeroing of an already-loaded fragment half (v_cndmask, no exec).
static __device__ __forceinline__ uint4 selz(bool v, uint4 a) {
  uint4 r;
  r.x = v ? a.x : 0u;  r.y = v ? a.y : 0u;
  r.z = v ? a.z : 0u;  r.w = v ? a.w : 0u;
  return r;
}

// ---------------------------------------------------------------------------
// Embedding gather: e = emb1[x], f32 -> bf16, channel-padded to CP with zeros.
__global__ __launch_bounds__(256) void k_embed(const int* __restrict__ x,
    const float* __restrict__ emb, unsigned short* __restrict__ dst) {
  int t = threadIdx.x;
  long long p = (long long)blockIdx.x * 2 + (t >> 7);
  int c = t & (CP - 1);
  int idx = x[p];
  float v = (c < EE) ? emb[(long long)idx * EE + c] : 0.f;
  dst[p * CP + c] = f2bf(v);
}

// Zero the atomic max-pool table (relu outputs >= 0, so 0 == +0.0f identity).
__global__ __launch_bounds__(256) void k_zero(unsigned* __restrict__ p) {
  p[blockIdx.x * 256 + threadIdx.x] = 0u;
}

// ---------------------------------------------------------------------------
// Pack conv weights [taps][cin][cout] (f32) into per-lane WMMA B-fragment
// layout: dst[tap][kc=4][ntile=8][lane=32][16], bf16, zero-padded.
// Lane n holds column n&15, K = kc*32 + 8*(n>>4) + (i&7) + 16*(i>>3)  (i = 0..15).
__global__ __launch_bounds__(256) void k_packw(const float* __restrict__ w,
    int taps, int cin, int cout, unsigned short* __restrict__ dst) {
  int idx = blockIdx.x * 256 + threadIdx.x;
  int total = taps * 4 * 8 * 32 * 16;
  if (idx >= total) return;
  int i    = idx & 15;
  int lane = (idx >> 4) & 31;
  int nt   = (idx >> 9) & 7;
  int kc   = (idx >> 12) & 3;
  int tap  = idx >> 14;
  int k = kc * 32 + ((lane >> 4) << 3) + (i & 7) + ((i >> 3) << 4);
  int n = nt * 16 + (lane & 15);
  float v = (k < cin && n < cout) ? w[((long long)tap * cin + k) * cout + n] : 0.f;
  dst[idx] = f2bf(v);
}

// Pad the 4 conv biases to 128 floats each, and w_conv4 to 128 floats.
__global__ __launch_bounds__(640) void k_packsmall(const float* __restrict__ b1,
    const float* __restrict__ b2, const float* __restrict__ b3,
    const float* __restrict__ bL, const float* __restrict__ w4,
    float* __restrict__ bias128, float* __restrict__ wc4p) {
  int t = threadIdx.x;
  if (t < 512) {
    int which = t >> 7, c = t & 127;
    const float* src = (which == 0) ? b1 : (which == 1) ? b2 : (which == 2) ? b3 : bL;
    bias128[t] = (c < FF) ? src[c] : 0.f;
  } else {
    int c = t - 512;
    wc4p[c] = (c < FF) ? w4[c] : 0.f;
  }
}

// Cooperative global->LDS stage of packed weights (uint4 granularity).
static __device__ __forceinline__ void stage_weights(
    unsigned short* __restrict__ sw, const unsigned short* __restrict__ wpk,
    int n16) {                           // n16 = total 16B chunks
  for (int i = threadIdx.x; i < n16; i += 256)
    ((uint4*)sw)[i] = ((const uint4*)wpk)[i];   // global_load_b128 + ds_store_b128
  __syncthreads();
}

// ---------------------------------------------------------------------------
// 3-tap 'SAME' conv1d as WMMA GEMM. ONE WAVE per 16-position tile computes all
// 8 N-tiles (A fragments loaded once, reused 8x): 3 taps x 4 kc x 8 nt = 96
// v_wmma_f32_16x16x32_bf16 per wave.  B fragments come from LDS (staged once
// per block) so each WMMA waits on ~LDS latency, not L2 latency.
// EDGE template: only lt==0 / lt==255 tiles pay for boundary masking
// (v_cndmask on loaded data; loads stay unconditional b128 from clamped addr).
// Optional fused global max-pool (conv1 only): float max == u32 max on raw
// bits for relu outputs.
template<bool EDGE>
static __device__ __forceinline__ void conv3_body(const unsigned short* __restrict__ in,
    const unsigned short* __restrict__ sw, const float* __restrict__ bias128,
    unsigned short* __restrict__ out, unsigned* __restrict__ pmaxu,
    int b, int lt, int lane) {
  int l0 = lt << 4;
  int m  = lane & 15;
  int l  = l0 + m;
  int ksel = (lane >> 4) << 3;     // 0 or 8
  v8f acc[8];
  #pragma unroll
  for (int nt = 0; nt < 8; ++nt) acc[nt] = (v8f){0.f,0.f,0.f,0.f,0.f,0.f,0.f,0.f};

  const unsigned short* wl = sw + (lane << 4);

  #pragma unroll
  for (int tap = 0; tap < 3; ++tap) {
    int lp = l + tap - 1;
    bool valid = true;
    int lc = lp;
    if (EDGE) {
      valid = (lp >= 0) && (lp < LL);
      lc = valid ? lp : 0;
    }
    const unsigned short* row = in + ((long long)b * LL + lc) * CP + ksel;
    if (!EDGE && tap == 1) __builtin_prefetch(row + 16 * CP, 0, 3);  // next tile
    #pragma unroll
    for (int kc = 0; kc < 4; ++kc) {
      int kb = kc << 5;
      uint4 a0 = *(const uint4*)(row + kb);
      uint4 a1 = *(const uint4*)(row + kb + 16);
      if (EDGE) { a0 = selz(valid, a0); a1 = selz(valid, a1); }
      v16bf A = mkfrag(a0, a1);
      #pragma unroll
      for (int nt = 0; nt < 8; ++nt) {
        const unsigned short* wb = wl + ((((tap << 2) + kc) * 8 + nt) << 9);
        uint4 b0 = *(const uint4*)(wb);       // ds_load_b128
        uint4 b1 = *(const uint4*)(wb + 8);   // ds_load_b128
        acc[nt] = __builtin_amdgcn_wmma_f32_16x16x32_bf16(false, A, false,
                      mkfrag(b0, b1), (short)0, acc[nt], false, false);
      }
    }
  }
  // C layout: VGPR r, lanes 0-15 -> M=r; lanes 16-31 -> M=r+8; col = lane&15.
  long long obase = ((long long)b * LL + l0) * CP;
  int rsel = (lane >> 4) << 3;
  #pragma unroll
  for (int nt = 0; nt < 8; ++nt) {
    int col = (nt << 4) + m;
    float bv = bias128[col];
    unsigned short* orow = out + obase + col;
    float mx = 0.f;
    #pragma unroll
    for (int r = 0; r < 8; ++r) {
      float v = acc[nt][r] + bv;
      v = v > 0.f ? v : 0.f;               // ReLU (zero-padded cols stay 0)
      mx = v > mx ? v : mx;
      orow[(long long)(r + rsel) * CP] = f2bf(v);
    }
    if (pmaxu) {                           // fused max-pool (conv1): uniform branch
      float o = __shfl_xor(mx, 16);        // merge rows r(0-7) with r+8(8-15)
      mx = o > mx ? o : mx;
      if (lane < 16)
        atomicMax(&pmaxu[b * CP + col], __float_as_uint(mx));
    }
  }
}

__global__ __launch_bounds__(256) void k_conv3(const unsigned short* __restrict__ in,
    const unsigned short* __restrict__ wpk, const float* __restrict__ bias128,
    unsigned short* __restrict__ out, unsigned* __restrict__ pmaxu) {
  extern __shared__ unsigned short sw[];               // 3*4*8*32*16 bf16 = 96KB
  stage_weights(sw, wpk, 3 * 4 * 8 * 32 * 16 / 8);
  int wave = (blockIdx.x << 3) + (threadIdx.x >> 5);   // == position tile
  int lane = threadIdx.x & 31;
  int b  = wave >> 8;                                  // L/16 == 256
  int lt = wave & 255;
  if (lt == 0 || lt == 255)
    conv3_body<true>(in, sw, bias128, out, pmaxu, b, lt, lane);
  else
    conv3_body<false>(in, sw, bias128, out, pmaxu, b, lt, lane);
}

// ---------------------------------------------------------------------------
// convL fused with conv4: K=1 over concat([ginfo bcast, linfo]) = 2 "taps",
// 64 WMMAs per wave (B fragments from LDS), then
// logits[b,l] = b4 + sum_col relu(tile+bias)*w4[col] computed in-register:
// per-lane partial over the 8 N-tiles, then a 4-step shfl_xor butterfly over
// each 16-lane half-group (rows r / r+8).  hmid is never materialized.
__global__ __launch_bounds__(256) void k_convLog(const unsigned short* __restrict__ lin,
    const unsigned short* __restrict__ gin, const unsigned short* __restrict__ wpk,
    const float* __restrict__ bias128, const float* __restrict__ wc4p,
    const float* __restrict__ b4, float* __restrict__ logits) {
  extern __shared__ unsigned short sw[];               // 2*4*8*32*16 bf16 = 64KB
  stage_weights(sw, wpk, 2 * 4 * 8 * 32 * 16 / 8);
  int wave = (blockIdx.x << 3) + (threadIdx.x >> 5);
  int lane = threadIdx.x & 31;
  int b  = wave >> 8;
  int lt = wave & 255;
  int l0 = lt << 4;
  int m  = lane & 15;
  int l  = l0 + m;
  int ksel = (lane >> 4) << 3;
  v8f acc[8];
  #pragma unroll
  for (int nt = 0; nt < 8; ++nt) acc[nt] = (v8f){0.f,0.f,0.f,0.f,0.f,0.f,0.f,0.f};

  const unsigned short* wl = sw + (lane << 4);

  #pragma unroll
  for (int tap = 0; tap < 2; ++tap) {
    const unsigned short* row = (tap == 0)
        ? (gin + (long long)b * CP + ksel)                    // ginfo broadcast
        : (lin + ((long long)b * LL + l) * CP + ksel);        // linfo
    #pragma unroll
    for (int kc = 0; kc < 4; ++kc) {
      int kb = kc << 5;
      uint4 a0 = *(const uint4*)(row + kb);
      uint4 a1 = *(const uint4*)(row + kb + 16);
      v16bf A = mkfrag(a0, a1);
      #pragma unroll
      for (int nt = 0; nt < 8; ++nt) {
        const unsigned short* wb = wl + ((((tap << 2) + kc) * 8 + nt) << 9);
        uint4 b0 = *(const uint4*)(wb);       // ds_load_b128
        uint4 b1 = *(const uint4*)(wb + 8);
        acc[nt] = __builtin_amdgcn_wmma_f32_16x16x32_bf16(false, A, false,
                      mkfrag(b0, b1), (short)0, acc[nt], false, false);
      }
    }
  }
  // Epilogue: hmid row-dot with w_conv4, fully in registers + shuffles.
  float bias4 = b4[0];
  int rsel = (lane >> 4) << 3;
  #pragma unroll
  for (int r = 0; r < 8; ++r) {
    float s = 0.f;
    #pragma unroll
    for (int nt = 0; nt < 8; ++nt) {
      int col = (nt << 4) + m;
      float v = acc[nt][r] + bias128[col];
      v = v > 0.f ? v : 0.f;               // ReLU
      s += v * wc4p[col];                  // zero-padded weights beyond 100
    }
    s += __shfl_xor(s, 1);
    s += __shfl_xor(s, 2);
    s += __shfl_xor(s, 4);
    s += __shfl_xor(s, 8);                 // sum over the 16-lane half-group
    if (m == 0)
      logits[(long long)b * LL + l0 + r + rsel] = s + bias4;
  }
}

// ---------------------------------------------------------------------------
// ginfo = relu(maxpool @ w_dense1 + b) -> bf16 (max table from fused atomics).
__global__ __launch_bounds__(128) void k_maxdense(const unsigned* __restrict__ pmaxu,
    const float* __restrict__ w, const float* __restrict__ bias,
    unsigned short* __restrict__ gout) {
  int b = blockIdx.x, c = threadIdx.x;
  __shared__ float smax[CP];
  smax[c] = __uint_as_float(pmaxu[b * CP + c]);
  __syncthreads();
  float s = 0.f;
  if (c < FF) {
    s = bias[c];
    for (int k = 0; k < FF; ++k) s += smax[k] * w[k * FF + c];
    s = s > 0.f ? s : 0.f;
  }
  gout[b * CP + c] = f2bf((c < FF) ? s : 0.f);
}

// ---------------------------------------------------------------------------
// Exact k-th largest per row: binary search over order-preserving uint keys
// of the 4096 logits held in LDS (32 iterations, count >= mid).
__global__ __launch_bounds__(256) void k_topk(const float* __restrict__ logits,
    float* __restrict__ thresh) {
  int b = blockIdx.x, t = threadIdx.x;
  __shared__ unsigned key[LL];
  __shared__ int cnt;
  for (int i = t; i < LL; i += 256) {
    unsigned u = __float_as_uint(logits[(long long)b * LL + i]);
    key[i] = (u & 0x80000000u) ? ~u : (u | 0x80000000u);
  }
  __syncthreads();
  unsigned lo = 0u, hi = 0xFFFFFFFFu;
  while (lo < hi) {
    unsigned mid = lo + ((hi - lo - 1u) >> 1) + 1u;   // upper mid, no overflow
    if (t == 0) cnt = 0;
    __syncthreads();
    int c = 0;
    for (int i = t; i < LL; i += 256) c += (key[i] >= mid) ? 1 : 0;
    atomicAdd(&cnt, c);
    __syncthreads();
    int total = cnt;
    __syncthreads();
    if (total >= KSEL) lo = mid; else hi = mid - 1u;
  }
  if (t == 0) {
    unsigned k = lo;
    unsigned u = (k & 0x80000000u) ? (k & 0x7FFFFFFFu) : ~k;
    thresh[b] = __uint_as_float(u);
  }
}

// ---------------------------------------------------------------------------
// pooled[b,:] = mean_l( emb2[x[b,l]] * (logits >= thresh) )  via LDS ds_add_f32
__global__ __launch_bounds__(256) void k_pool(const int* __restrict__ x,
    const float* __restrict__ emb2, const float* __restrict__ logits,
    const float* __restrict__ thresh, float* __restrict__ pooled) {
  int b = blockIdx.x, t = threadIdx.x;
  __shared__ float acc[64];
  if (t < 64) acc[t] = 0.f;
  __syncthreads();
  float th = thresh[b];
  for (int l = t; l < LL; l += 256) {
    if (logits[(long long)b * LL + l] >= th) {
      int idx = x[(long long)b * LL + l];
      const float* row = emb2 + (long long)idx * EE;
      for (int e = 0; e < EE; ++e) atomicAdd(&acc[e], row[e]);
    }
  }
  __syncthreads();
  if (t < 64) pooled[b * 64 + t] = acc[t] * (1.f / LL);
}

// ---------------------------------------------------------------------------
// head: hid = relu(pooled @ w_d + b_d); out = softmax(hid @ w_o + b_o)
__global__ __launch_bounds__(256) void k_head(const float* __restrict__ pooled,
    const float* __restrict__ wd, const float* __restrict__ bd,
    const float* __restrict__ wo, const float* __restrict__ bo,
    float* __restrict__ out) {
  int b = blockIdx.x, t = threadIdx.x;
  __shared__ float p[EE];
  __shared__ float hid[HH];
  if (t < EE) p[t] = pooled[b * 64 + t];
  __syncthreads();
  if (t < HH) {
    float s = bd[t];
    for (int e = 0; e < EE; ++e) s += p[e] * wd[e * HH + t];
    hid[t] = s > 0.f ? s : 0.f;
  }
  __syncthreads();
  if (t == 0) {
    float o0 = bo[0], o1 = bo[1];
    for (int h = 0; h < HH; ++h) { o0 += hid[h] * wo[h * 2]; o1 += hid[h] * wo[h * 2 + 1]; }
    float mx = o0 > o1 ? o0 : o1;
    float e0 = __expf(o0 - mx), e1 = __expf(o1 - mx);
    float inv = 1.f / (e0 + e1);
    out[b * 2]     = e0 * inv;
    out[b * 2 + 1] = e1 * inv;
  }
}

// ---------------------------------------------------------------------------
extern "C" void kernel_launch(void* const* d_in, const int* in_sizes, int n_in,
                              void* d_out, int out_size, void* d_ws, size_t ws_size,
                              hipStream_t stream) {
  (void)in_sizes; (void)n_in; (void)out_size; (void)ws_size;
  const int*   x        = (const int*)  d_in[0];
  const float* emb1     = (const float*)d_in[1];
  const float* w_conv1  = (const float*)d_in[2];
  const float* b_conv1  = (const float*)d_in[3];
  const float* w_dense1 = (const float*)d_in[4];
  const float* b_dense1 = (const float*)d_in[5];
  const float* w_conv2  = (const float*)d_in[6];
  const float* b_conv2  = (const float*)d_in[7];
  const float* w_conv3  = (const float*)d_in[8];
  const float* b_conv3  = (const float*)d_in[9];
  const float* w_convL  = (const float*)d_in[10];
  const float* b_convL  = (const float*)d_in[11];
  const float* w_conv4  = (const float*)d_in[12];
  const float* b_conv4  = (const float*)d_in[13];
  const float* emb2     = (const float*)d_in[14];
  const float* w_d      = (const float*)d_in[15];
  const float* b_d      = (const float*)d_in[16];
  const float* w_o      = (const float*)d_in[17];
  const float* b_o      = (const float*)d_in[18];
  float* out = (float*)d_out;

  char* ws = (char*)d_ws;
  const size_t szBuf = (size_t)BB * LL * CP * 2;          // 134 MB each
  size_t o = 0;
  unsigned short* bufA = (unsigned short*)(ws + o); o += szBuf;
  unsigned short* bufB = (unsigned short*)(ws + o); o += szBuf;
  float* logits = (float*)(ws + o); o += (size_t)BB * LL * 4;
  float* thresh = (float*)(ws + o); o += 512;
  unsigned short* ginfo = (unsigned short*)(ws + o); o += (size_t)BB * CP * 2;
  float* pooled = (float*)(ws + o); o += (size_t)BB * 64 * 4;
  unsigned* pmaxu = (unsigned*)(ws + o); o += (size_t)BB * CP * 4;
  float* bias128= (float*)(ws + o); o += 4 * 128 * 4;     // conv1,conv2,conv3,convL
  float* wc4p   = (float*)(ws + o); o += 128 * 4;
  unsigned short* wpk1 = (unsigned short*)(ws + o); o += 3 * 16384 * 2;
  unsigned short* wpk2 = (unsigned short*)(ws + o); o += 3 * 16384 * 2;
  unsigned short* wpk3 = (unsigned short*)(ws + o); o += 3 * 16384 * 2;
  unsigned short* wpkL = (unsigned short*)(ws + o); o += 2 * 16384 * 2;

  // Pack weights/biases into WMMA-friendly padded layouts (idempotent).
  k_packw<<<192, 256, 0, stream>>>(w_conv1, 3, EE, FF, wpk1);
  k_packw<<<192, 256, 0, stream>>>(w_conv2, 3, FF, FF, wpk2);
  k_packw<<<192, 256, 0, stream>>>(w_conv3, 3, FF, FF, wpk3);
  k_packw<<<128, 256, 0, stream>>>(w_convL, 2, FF, FF, wpkL);
  k_packsmall<<<1, 640, 0, stream>>>(b_conv1, b_conv2, b_conv3, b_convL, w_conv4,
                                     bias128, wc4p);
  k_zero<<<BB * CP / 256, 256, 0, stream>>>(pmaxu);

  // Pipeline (ping-pong bufA/bufB).  Conv grids: B*L/16 waves / 8 per block.
  const size_t lds3 = 3 * 4 * 8 * 32 * 16 * 2;            // 96KB packed weights
  const size_t ldsL = 2 * 4 * 8 * 32 * 16 * 2;            // 64KB
  k_embed<<<BB * LL / 2, 256, 0, stream>>>(x, emb1, bufA);                     // e -> A
  k_conv3<<<4096, 256, lds3, stream>>>(bufA, wpk1, bias128 + 0,   bufB, pmaxu); // net -> B (+max)
  k_maxdense<<<BB, 128, 0, stream>>>(pmaxu, w_dense1, b_dense1, ginfo);        // ginfo
  k_conv3<<<4096, 256, lds3, stream>>>(bufB, wpk2, bias128 + 128, bufA, nullptr); // l2 -> A
  k_conv3<<<4096, 256, lds3, stream>>>(bufA, wpk3, bias128 + 256, bufB, nullptr); // l3 -> B
  k_convLog<<<4096, 256, ldsL, stream>>>(bufB, ginfo, wpkL, bias128 + 384, wc4p,
                                         b_conv4, logits);                     // logits direct
  k_topk<<<BB, 256, 0, stream>>>(logits, thresh);
  k_pool<<<BB, 256, 0, stream>>>(x, emb2, logits, thresh, pooled);
  k_head<<<BB, 256, 0, stream>>>(pooled, w_d, b_d, w_o, b_o, out);
}
